// GroupedQueryAttention_73512660238620
// MI455X (gfx1250) — compile-verified, matrix operand order checked by reading, not run on hardware
//
#include <hip/hip_runtime.h>

typedef __attribute__((ext_vector_type(16))) __bf16 v16bf;
typedef __attribute__((ext_vector_type(8)))  float  v8f;

#define T_SEQ  2048
#define DMODEL 2048
#define N_HQ   32
#define N_HKV  8
#define DHEAD  64
#define LDSW   72   // LDS row stride in ushorts: 144B, 16B-aligned, bank-conflict-free

#if defined(__gfx1250__) && __has_builtin(__builtin_amdgcn_global_load_async_to_lds_b128)
#define USE_ASYNC 1
#else
#define USE_ASYNC 0
#endif

#if USE_ASYNC
// Builtin signature (from clang diagnostic): (v4i AS1*, v4i AS3*, imm offset, imm cpol)
typedef int v4i_gcc __attribute__((vector_size(16)));
typedef v4i_gcc __attribute__((address_space(1))) gv4i;   // global int4
typedef v4i_gcc __attribute__((address_space(3))) lv4i;   // LDS int4
#if __has_builtin(__builtin_amdgcn_s_wait_asynccnt)
#define WAIT_ASYNC(n) __builtin_amdgcn_s_wait_asynccnt(n)
#else
#define WAIT_ASYNC(n) asm volatile("s_wait_asynccnt %0" ::"i"(n) : "memory")
#endif
#else
#define WAIT_ASYNC(n)
#endif

__device__ __forceinline__ unsigned short f2bf(float f) {
  unsigned int u = __builtin_bit_cast(unsigned int, f);
  u += 0x7FFFu + ((u >> 16) & 1u);          // round-to-nearest-even
  return (unsigned short)(u >> 16);
}

union FragBF { v16bf v; unsigned int u[8]; };

__device__ __forceinline__ v8f v8f_zero() {
  v8f z = {0.f, 0.f, 0.f, 0.f, 0.f, 0.f, 0.f, 0.f};
  return z;
}

// ---------------------------------------------------------------- casts
__global__ void cast_bf16_kernel(const float* __restrict__ in,
                                 unsigned short* __restrict__ out, int n) {
  int i = blockIdx.x * blockDim.x + threadIdx.x;
  if (i < n) out[i] = f2bf(in[i]);
}

// RoPE (interleaved pairs) + cast to bf16.  X: (T, H, DH) f32
__global__ void rope_cast_kernel(const float* __restrict__ X,
                                 const float* __restrict__ cosT,
                                 const float* __restrict__ sinT,
                                 unsigned short* __restrict__ out,
                                 int H, int n /* T*H*DH/2 */) {
  int i = blockIdx.x * blockDim.x + threadIdx.x;
  if (i >= n) return;
  int j = i & 31;                 // DH/2 = 32
  int h = (i >> 5) % H;
  int t = i / (32 * H);
  int base = (t * H + h) * DHEAD + 2 * j;
  float x0 = X[base], x1 = X[base + 1];
  float c = cosT[t * 32 + j], s = sinT[t * 32 + j];
  out[base]     = f2bf(x0 * c - x1 * s);
  out[base + 1] = f2bf(x0 * s + x1 * c);
}

// V: (T, HKV*DH) f32  ->  Vt: (HKV*DH, T) bf16   (keys contiguous)
__global__ void vtranspose_kernel(const float* __restrict__ V,
                                  unsigned short* __restrict__ Vt) {
  int i = blockIdx.x * blockDim.x + threadIdx.x;
  if (i >= T_SEQ * N_HKV * DHEAD) return;
  int col = i & (N_HKV * DHEAD - 1);   // 512
  int t   = i >> 9;
  Vt[(size_t)col * T_SEQ + t] = f2bf(V[i]);
}

// ---------------------------------------------------------------- GEMM
// Stage one 64-wide K slab (A: 128 rows, B: 64 rows) into LDS.
// Async path: GLOBAL_LOAD_ASYNC_TO_LDS_B128 (ASYNCcnt), 6 ops/thread/slab.
__device__ __forceinline__ void stage_slab(
    const unsigned short* __restrict__ A, const unsigned short* __restrict__ B,
    unsigned short (*As)[LDSW], unsigned short (*Bs)[LDSW],
    int bm0, int bn0, int K, int k0, int tid) {
  {   // A slab: 128 rows x 64 ushorts; per thread 4 x 16B
    int r  = tid >> 1;
    int kc = (tid & 1) * 32;
    const unsigned short* g = A + (size_t)(bm0 + r) * K + k0 + kc;
    unsigned short* l = &As[r][kc];
#if USE_ASYNC
#pragma unroll
    for (int q = 0; q < 4; ++q)
      __builtin_amdgcn_global_load_async_to_lds_b128(
          (gv4i*)(g + 8 * q), (lv4i*)(l + 8 * q), 0, 0);
#else
    const uint4* gp = (const uint4*)g;
    uint4* lp = (uint4*)l;
#pragma unroll
    for (int q = 0; q < 4; ++q) lp[q] = gp[q];
#endif
  }
  {   // B slab: 64 rows x 64 ushorts; per thread 2 x 16B
    int r  = tid >> 2;
    int kc = (tid & 3) * 16;
    const unsigned short* g = B + (size_t)(bn0 + r) * K + k0 + kc;
    unsigned short* l = &Bs[r][kc];
#if USE_ASYNC
#pragma unroll
    for (int q = 0; q < 2; ++q)
      __builtin_amdgcn_global_load_async_to_lds_b128(
          (gv4i*)(g + 8 * q), (lv4i*)(l + 8 * q), 0, 0);
#else
    const uint4* gp = (const uint4*)g;
    uint4* lp = (uint4*)l;
#pragma unroll
    for (int q = 0; q < 2; ++q) lp[q] = gp[q];
#endif
  }
}

// C(MxN,f32) = A(MxK,bf16 row-major) * B^T  where B given as (NxK,bf16 row-major)
// block: 256 thr = 8 waves; block tile 128x64; wave tile 32x32; K-step 64,
// double-buffered LDS with async prefetch of the next slab during compute.
__global__ __launch_bounds__(256) void gemm_bf16_wmma(
    const unsigned short* __restrict__ A, const unsigned short* __restrict__ B,
    float* __restrict__ C, int M, int N, int K) {
  __shared__ __align__(16) unsigned short As[2][128][LDSW];
  __shared__ __align__(16) unsigned short Bs[2][64][LDSW];
  const int tid   = threadIdx.x;
  const int lane  = tid & 31;
  const int wave  = tid >> 5;
  const int waveM = wave >> 1;            // 0..3
  const int waveN = wave & 1;             // 0..1
  const int l15   = lane & 15;
  const int hi8   = (lane >> 4) << 3;     // 0 or 8 (C-layout row offset)
  const int kOff  = hi8;                  // A/B-layout K offset
  const int bm0 = blockIdx.y * 128;
  const int bn0 = blockIdx.x * 64;

  v8f acc[2][2];
#pragma unroll
  for (int i = 0; i < 2; ++i)
#pragma unroll
    for (int j = 0; j < 2; ++j) acc[i][j] = v8f_zero();

  stage_slab(A, B, As[0], Bs[0], bm0, bn0, K, 0, tid);

  const int nIter = K >> 6;               // K / 64
  for (int it = 0; it < nIter; ++it) {
    const int buf = it & 1;
    if (it + 1 < nIter) {
      stage_slab(A, B, As[buf ^ 1], Bs[buf ^ 1], bm0, bn0, K, (it + 1) * 64, tid);
      WAIT_ASYNC(6);                      // current slab done; prefetch in flight
    } else {
      WAIT_ASYNC(0);
    }
    __syncthreads();

#pragma unroll
    for (int kk = 0; kk < 64; kk += 32) {
      FragBF a[2], b[2];
#pragma unroll
      for (int i = 0; i < 2; ++i) {
        int row = waveM * 32 + i * 16 + l15;
#pragma unroll
        for (int v = 0; v < 8; ++v) {
          int kf = 2 * v + kOff + ((v >= 4) ? 8 : 0);
          a[i].u[v] = *(const unsigned int*)&As[buf][row][kk + kf];
        }
      }
#pragma unroll
      for (int j = 0; j < 2; ++j) {
        int row = waveN * 32 + j * 16 + l15;
#pragma unroll
        for (int v = 0; v < 8; ++v) {
          int kf = 2 * v + kOff + ((v >= 4) ? 8 : 0);
          b[j].u[v] = *(const unsigned int*)&Bs[buf][row][kk + kf];
        }
      }
#pragma unroll
      for (int i = 0; i < 2; ++i)
#pragma unroll
        for (int j = 0; j < 2; ++j)
          acc[i][j] = __builtin_amdgcn_wmma_f32_16x16x32_bf16(
              false, a[i].v, false, b[j].v, (short)0, acc[i][j], false, false);
    }
    __syncthreads();
  }

#pragma unroll
  for (int i = 0; i < 2; ++i)
#pragma unroll
    for (int j = 0; j < 2; ++j)
#pragma unroll
      for (int v = 0; v < 8; ++v) {
        int row = bm0 + waveM * 32 + i * 16 + v + hi8;
        int col = bn0 + waveN * 32 + j * 16 + l15;
        C[(size_t)row * N + col] = acc[i][j][v];
      }
}

// ---------------------------------------------------------------- attention
// Per wave: one q-head, one 16-row query tile, flash (online softmax).
// Q:(T,HQ,DH) bf16  K:(T,HKV,DH) bf16  Vt:(HKV*DH,T) bf16  O:(T,HQ*DH) bf16
__global__ __launch_bounds__(128) void attn_wmma(
    const unsigned short* __restrict__ Q, const unsigned short* __restrict__ Kc,
    const unsigned short* __restrict__ Vt, unsigned short* __restrict__ O) {
  __shared__ unsigned short P[4][16][34];   // per-wave P tile (16 q x 32 keys)
  const int tid  = threadIdx.x;
  const int lane = tid & 31;
  const int w    = tid >> 5;
  const int l15  = lane & 15;
  const int hi8  = (lane >> 4) << 3;
  const int kOff = hi8;
  const int qh   = blockIdx.x;
  const int kvh  = qh >> 2;                 // NREP = 4
  const int qr0  = blockIdx.y * 64 + w * 16;
  const float NEG = -3.0e38f;

  // Q fragments (dh halves 0..31, 32..63), resident for whole kernel
  FragBF qf[2];
#pragma unroll
  for (int h = 0; h < 2; ++h) {
    const unsigned short* qrow =
        Q + ((size_t)(qr0 + l15) * N_HQ + qh) * DHEAD + 32 * h;
#pragma unroll
    for (int v = 0; v < 8; ++v) {
      int kf = 2 * v + kOff + ((v >= 4) ? 8 : 0);
      qf[h].u[v] = *(const unsigned int*)(qrow + kf);
    }
  }

  v8f o[4];
#pragma unroll
  for (int t4 = 0; t4 < 4; ++t4) o[t4] = v8f_zero();
  float mrow[8], lrow[8];
#pragma unroll
  for (int v = 0; v < 8; ++v) { mrow[v] = NEG; lrow[v] = 0.f; }

  const int nkb = (qr0 + 15) / 32 + 1;      // causal: keys 0 .. qr0+15
  for (int kb = 0; kb < nkb; ++kb) {
    const int kb0 = kb * 32;

    // K fragments: 2 key subtiles x 2 dh halves (B-matrix of Q*K^T)
    FragBF kfr[2][2];
#pragma unroll
    for (int s = 0; s < 2; ++s) {
      const unsigned short* krow =
          Kc + ((size_t)(kb0 + s * 16 + l15) * N_HKV + kvh) * DHEAD;
#pragma unroll
      for (int h = 0; h < 2; ++h)
#pragma unroll
        for (int v = 0; v < 8; ++v) {
          int kf = 2 * v + kOff + ((v >= 4) ? 8 : 0);
          kfr[s][h].u[v] = *(const unsigned int*)(krow + 32 * h + kf);
        }
    }

    v8f s0 = v8f_zero(), s1 = v8f_zero();
    s0 = __builtin_amdgcn_wmma_f32_16x16x32_bf16(false, qf[0].v, false, kfr[0][0].v, (short)0, s0, false, false);
    s0 = __builtin_amdgcn_wmma_f32_16x16x32_bf16(false, qf[1].v, false, kfr[0][1].v, (short)0, s0, false, false);
    s1 = __builtin_amdgcn_wmma_f32_16x16x32_bf16(false, qf[0].v, false, kfr[1][0].v, (short)0, s1, false, false);
    s1 = __builtin_amdgcn_wmma_f32_16x16x32_bf16(false, qf[1].v, false, kfr[1][1].v, (short)0, s1, false, false);

    const bool needMask = (kb0 + 31 > qr0);
#pragma unroll
    for (int v = 0; v < 8; ++v) {
      float a0 = s0[v] * 0.125f;            // 1/sqrt(64)
      float a1 = s1[v] * 0.125f;
      if (needMask) {
        int qrow = qr0 + v + hi8;
        if (kb0 + l15 > qrow)      a0 = NEG;
        if (kb0 + 16 + l15 > qrow) a1 = NEG;
      }
      float mx = fmaxf(a0, a1);             // row max over 16-lane group
      mx = fmaxf(mx, __shfl_xor(mx, 1));
      mx = fmaxf(mx, __shfl_xor(mx, 2));
      mx = fmaxf(mx, __shfl_xor(mx, 4));
      mx = fmaxf(mx, __shfl_xor(mx, 8));
      float mn = fmaxf(mrow[v], mx);
      float es = __expf(mrow[v] - mn);
      mrow[v] = mn;
      float p0 = __expf(a0 - mn);
      float p1 = __expf(a1 - mn);
      float rs = p0 + p1;
      rs += __shfl_xor(rs, 1);
      rs += __shfl_xor(rs, 2);
      rs += __shfl_xor(rs, 4);
      rs += __shfl_xor(rs, 8);
      lrow[v] = lrow[v] * es + rs;
      o[0][v] *= es; o[1][v] *= es; o[2][v] *= es; o[3][v] *= es;
      P[w][v + hi8][l15]      = f2bf(p0);   // C-layout -> LDS tile
      P[w][v + hi8][l15 + 16] = f2bf(p1);
    }

    asm volatile("" ::: "memory");          // DS is in-order per wave; block reordering
    FragBF pf;                              // re-read P as A-layout fragment
#pragma unroll
    for (int v = 0; v < 8; ++v) {
      int kf = 2 * v + kOff + ((v >= 4) ? 8 : 0);
      pf.u[v] = *(const unsigned int*)&P[w][l15][kf];
    }
    asm volatile("" ::: "memory");

#pragma unroll
    for (int t4 = 0; t4 < 4; ++t4) {        // O += P * V, N tiles over DH
      FragBF vf;
      const unsigned short* vtrow =
          Vt + ((size_t)(kvh * DHEAD + t4 * 16 + l15)) * T_SEQ + kb0;
#pragma unroll
      for (int v = 0; v < 8; ++v) {
        int kf = 2 * v + kOff + ((v >= 4) ? 8 : 0);
        vf.u[v] = *(const unsigned int*)(vtrow + kf);
      }
      o[t4] = __builtin_amdgcn_wmma_f32_16x16x32_bf16(
          false, pf.v, false, vf.v, (short)0, o[t4], false, false);
    }
  }

  // normalize + store directly in bf16 (feeds the Wo GEMM as its A matrix)
#pragma unroll
  for (int v = 0; v < 8; ++v) {
    float inv = 1.0f / lrow[v];
    int row = qr0 + v + hi8;
    unsigned short* orow = O + (size_t)row * DMODEL + qh * DHEAD;
#pragma unroll
    for (int t4 = 0; t4 < 4; ++t4) orow[t4 * 16 + l15] = f2bf(o[t4][v] * inv);
  }
}

// ---------------------------------------------------------------- launch
extern "C" void kernel_launch(void* const* d_in, const int* in_sizes, int n_in,
                              void* d_out, int out_size, void* d_ws, size_t ws_size,
                              hipStream_t stream) {
  (void)in_sizes; (void)n_in; (void)out_size; (void)ws_size;
  const float* x  = (const float*)d_in[0];
  const float* Wq = (const float*)d_in[1];
  const float* Wk = (const float*)d_in[2];
  const float* Wv = (const float*)d_in[3];
  const float* Wo = (const float*)d_in[4];
  const float* rc = (const float*)d_in[5];
  const float* rs = (const float*)d_in[6];
  // d_in[7] = causal mask -> applied analytically

  char* ws = (char*)d_ws;
  size_t off = 0;
  auto take = [&](size_t bytes) {
    void* p = ws + off;
    off += (bytes + 255) & ~(size_t)255;
    return p;
  };
  const size_t NXD = (size_t)T_SEQ * DMODEL;          // 4.19M
  const size_t NKV = (size_t)T_SEQ * N_HKV * DHEAD;   // 1.05M
  unsigned short* xb  = (unsigned short*)take(NXD * 2);
  unsigned short* Wqb = (unsigned short*)take(NXD * 2);
  unsigned short* Wkb = (unsigned short*)take(NKV * 2);
  unsigned short* Wvb = (unsigned short*)take(NKV * 2);
  unsigned short* Wob = (unsigned short*)take(NXD * 2);
  float*          Qf  = (float*)take(NXD * 4);
  float*          Kf  = (float*)take(NKV * 4);
  float*          Vf  = (float*)take(NKV * 4);
  unsigned short* Qb  = (unsigned short*)take(NXD * 2);
  unsigned short* Kb  = (unsigned short*)take(NKV * 2);
  unsigned short* Vtb = (unsigned short*)take(NKV * 2);
  unsigned short* AOb = (unsigned short*)take(NXD * 2);

  const int nxd = (int)NXD, nkv = (int)NKV;
  cast_bf16_kernel<<<(nxd + 255) / 256, 256, 0, stream>>>(x,  xb,  nxd);
  cast_bf16_kernel<<<(nxd + 255) / 256, 256, 0, stream>>>(Wq, Wqb, nxd);
  cast_bf16_kernel<<<(nkv + 255) / 256, 256, 0, stream>>>(Wk, Wkb, nkv);
  cast_bf16_kernel<<<(nkv + 255) / 256, 256, 0, stream>>>(Wv, Wvb, nkv);
  cast_bf16_kernel<<<(nxd + 255) / 256, 256, 0, stream>>>(Wo, Wob, nxd);

  dim3 gq(DMODEL / 64, T_SEQ / 128);                  // N/64, M/128
  dim3 gk((N_HKV * DHEAD) / 64, T_SEQ / 128);
  gemm_bf16_wmma<<<gq, 256, 0, stream>>>(xb, Wqb, Qf, T_SEQ, DMODEL, DMODEL);
  gemm_bf16_wmma<<<gk, 256, 0, stream>>>(xb, Wkb, Kf, T_SEQ, N_HKV * DHEAD, DMODEL);
  gemm_bf16_wmma<<<gk, 256, 0, stream>>>(xb, Wvb, Vf, T_SEQ, N_HKV * DHEAD, DMODEL);

  int nprQ = T_SEQ * N_HQ * 32, nprK = T_SEQ * N_HKV * 32;
  rope_cast_kernel<<<(nprQ + 255) / 256, 256, 0, stream>>>(Qf, rc, rs, Qb, N_HQ, nprQ);
  rope_cast_kernel<<<(nprK + 255) / 256, 256, 0, stream>>>(Kf, rc, rs, Kb, N_HKV, nprK);
  vtranspose_kernel<<<(nkv + 255) / 256, 256, 0, stream>>>(Vf, Vtb);

  dim3 ga(N_HQ, T_SEQ / 64);
  attn_wmma<<<ga, 128, 0, stream>>>(Qb, Kb, Vtb, AOb);

  gemm_bf16_wmma<<<gq, 256, 0, stream>>>(AOb, Wob, (float*)d_out, T_SEQ, DMODEL, DMODEL);
}
